// LSTMCell_20169166422057
// MI455X (gfx1250) — compile-verified, matrix operand order checked by reading, not run on hardware
//
#include <hip/hip_runtime.h>
#include <hip/hip_bf16.h>
#include <stdint.h>

// Problem dimensions (fixed by the reference)
#define BATCH 16384
#define IN    512     // INPUT_SIZE
#define H     512     // HIDDEN
#define OH    1536    // ORDER * HIDDEN
#define G4    2048    // 4 * HIDDEN
#define MASK  0.8f    // 1 - DROPMASK (eval mode)

typedef __attribute__((ext_vector_type(16))) __bf16       v16bf;
typedef __attribute__((ext_vector_type(8)))  float        v8f;
typedef __attribute__((ext_vector_type(8)))  unsigned int v8u;
typedef int v4i_ __attribute__((vector_size(16)));  // matches async-LDS builtin param

#define AS1 __attribute__((address_space(1)))
#define AS3 __attribute__((address_space(3)))

#if defined(__gfx1250__) && __has_builtin(__builtin_amdgcn_global_load_async_to_lds_b128) && \
    __has_builtin(__builtin_amdgcn_s_wait_asynccnt)
#define USE_ASYNC_LDS 1
#else
#define USE_ASYNC_LDS 0
#endif

__device__ __forceinline__ uint16_t f32_to_bf16_rne(float f) {
  uint32_t u = __builtin_bit_cast(uint32_t, f);
  u += 0x7FFFu + ((u >> 16) & 1u);
  return (uint16_t)(u >> 16);
}

__device__ __forceinline__ float sigmoidf_fast(float x) {
  return 1.0f / (1.0f + __expf(-x));
}

// Load a 16x32 bf16 operand tile into the CDNA5 WMMA per-lane layout.
// Per ISA 7.12.2 (16-bit A 16x32): lane L -> row L%16, K-half = L/16,
// VGPR v holds K pair { (v>=4?16:0) + (L/16)*8 + (v%4)*2 }.
// The two K-runs per lane are 16B-contiguous -> compiler emits 2x b128
// (global_load_b128 from global, ds_load_b128 from LDS).
__device__ __forceinline__ v16bf load_frag(const uint16_t* __restrict__ p, int ld) {
  const int lane = threadIdx.x & 31;
  const uint16_t* r = p + (size_t)(lane & 15) * ld + (lane >> 4) * 8;
  v8u t;
#pragma unroll
  for (int v = 0; v < 8; ++v) {
    const int k = ((v >> 2) << 4) + ((v & 3) << 1);  // 0,2,4,6,16,18,20,22
    t[v] = *reinterpret_cast<const uint32_t*>(r + k);
  }
  return __builtin_bit_cast(v16bf, t);
}

// Stage one 64x32 bf16 tile (global, row stride ldg) into LDS (row stride 32),
// 256 threads x 16B. Async (ASYNCcnt-tracked) when the toolchain exposes the
// gfx1250 builtin; otherwise synchronous load + ds_store fallback.
__device__ __forceinline__ void stage_tile(uint16_t* __restrict__ dstLds,
                                           const uint16_t* __restrict__ src, int ldg) {
  const int t   = threadIdx.x;
  const int row = t >> 2;        // 64 rows, 4 threads per row
  const int c8  = (t & 3) << 3;  // 8 bf16 = 16B per thread
  const uint16_t* g = src + (size_t)row * ldg + c8;
  uint16_t* l = dstLds + row * 32 + c8;
#if USE_ASYNC_LDS
  __builtin_amdgcn_global_load_async_to_lds_b128((AS1 v4i_*)g, (AS3 v4i_*)l, 0, 0);
#else
  *reinterpret_cast<uint4*>(l) = *reinterpret_cast<const uint4*>(g);
#endif
}

__device__ __forceinline__ void async_wait0() {
#if USE_ASYNC_LDS
  __builtin_amdgcn_s_wait_asynccnt(0);
#endif
}

// ---------------- conversion kernels ----------------

// 4 floats -> 4 bf16 per thread (f32 read as float4, bf16 written as uint2)
__global__ void cvt_bf16x4(const float4* __restrict__ src, uint2* __restrict__ dst, int n4) {
  int i = blockIdx.x * blockDim.x + threadIdx.x;
  if (i < n4) {
    float4 f = src[i];
    uint2 o;
    o.x = (uint32_t)f32_to_bf16_rne(f.x) | ((uint32_t)f32_to_bf16_rne(f.y) << 16);
    o.y = (uint32_t)f32_to_bf16_rne(f.z) | ((uint32_t)f32_to_bf16_rne(f.w) << 16);
    dst[i] = o;
  }
}

// src: row-major [K, N] f32  ->  dst: row-major [N, K] bf16
__global__ void cvt_transpose_bf16(const float* __restrict__ src, uint16_t* __restrict__ dst,
                                   int K, int N) {
  int i = blockIdx.x * blockDim.x + threadIdx.x;
  if (i < K * N) {
    int k = i / N;
    int n = i - k * N;
    dst[(size_t)n * K + k] = f32_to_bf16_rne(src[i]);
  }
}

// ---------------- GEMM 1: s = sigmoid(x@W_memi + me@W_memh + mem_bias) * me ----------------
// grid (BATCH/128, OH/64), block 256 (8 waves); wave tile 32x32 (2x2 frags),
// workgroup tile 128x64. 2x b128 per WMMA.

__global__ __launch_bounds__(256) void memgate_gemm(
    const uint16_t* __restrict__ xb, const uint16_t* __restrict__ meb,
    const uint16_t* __restrict__ WmemiT, const uint16_t* __restrict__ WmemhT,
    const float* __restrict__ mem_bias, const float* __restrict__ me_f32,
    uint16_t* __restrict__ sb) {
  const int w  = threadIdx.x >> 5;
  const int m0 = blockIdx.x * 128 + (w >> 1) * 32;
  const int n0 = blockIdx.y * 64 + (w & 1) * 32;

  v8f acc[2][2] = {};

  const uint16_t* a0 = xb + (size_t)m0 * IN;
  const uint16_t* b0 = WmemiT + (size_t)n0 * IN;
  for (int k = 0; k < IN; k += 32) {
    __builtin_prefetch(a0 + k + 64, 0, 3);
    v16bf fa0 = load_frag(a0 + k, IN);
    v16bf fa1 = load_frag(a0 + (size_t)16 * IN + k, IN);
    v16bf fb0 = load_frag(b0 + k, IN);
    v16bf fb1 = load_frag(b0 + (size_t)16 * IN + k, IN);
    acc[0][0] = __builtin_amdgcn_wmma_f32_16x16x32_bf16(false, fa0, false, fb0, (short)0, acc[0][0], false, false);
    acc[0][1] = __builtin_amdgcn_wmma_f32_16x16x32_bf16(false, fa0, false, fb1, (short)0, acc[0][1], false, false);
    acc[1][0] = __builtin_amdgcn_wmma_f32_16x16x32_bf16(false, fa1, false, fb0, (short)0, acc[1][0], false, false);
    acc[1][1] = __builtin_amdgcn_wmma_f32_16x16x32_bf16(false, fa1, false, fb1, (short)0, acc[1][1], false, false);
  }
  const uint16_t* a1 = meb + (size_t)m0 * OH;
  const uint16_t* b1 = WmemhT + (size_t)n0 * OH;
  for (int k = 0; k < OH; k += 32) {
    __builtin_prefetch(a1 + k + 64, 0, 3);
    v16bf fa0 = load_frag(a1 + k, OH);
    v16bf fa1 = load_frag(a1 + (size_t)16 * OH + k, OH);
    v16bf fb0 = load_frag(b1 + k, OH);
    v16bf fb1 = load_frag(b1 + (size_t)16 * OH + k, OH);
    acc[0][0] = __builtin_amdgcn_wmma_f32_16x16x32_bf16(false, fa0, false, fb0, (short)0, acc[0][0], false, false);
    acc[0][1] = __builtin_amdgcn_wmma_f32_16x16x32_bf16(false, fa0, false, fb1, (short)0, acc[0][1], false, false);
    acc[1][0] = __builtin_amdgcn_wmma_f32_16x16x32_bf16(false, fa1, false, fb0, (short)0, acc[1][0], false, false);
    acc[1][1] = __builtin_amdgcn_wmma_f32_16x16x32_bf16(false, fa1, false, fb1, (short)0, acc[1][1], false, false);
  }

  const int lane = threadIdx.x & 31;
#pragma unroll
  for (int mf = 0; mf < 2; ++mf) {
#pragma unroll
    for (int nf = 0; nf < 2; ++nf) {
      const int col = n0 + nf * 16 + (lane & 15);
      const float bc = mem_bias[col];
#pragma unroll
      for (int v = 0; v < 8; ++v) {
        const int row = m0 + mf * 16 + v + ((lane >> 4) << 3);
        const size_t idx = (size_t)row * OH + col;
        const float s = sigmoidf_fast(acc[mf][nf][v] + bc) * me_f32[idx];
        sb[idx] = f32_to_bf16_rne(s);
      }
    }
  }
}

// ---------------- GEMM 2: gates + t + fused LSTM epilogue ----------------
// grid (BATCH/64, H/64), block 256 (8 waves, 2M x 4N); wave tile 32x16 per matrix,
// 10 live f32 accumulators. A operands (x, h, then s) double-buffered in LDS via
// async global->LDS staging; B (weights) direct from global (L2/WGP$-resident).

__global__ __launch_bounds__(256) void lstm_gemm(
    const uint16_t* __restrict__ xb, const uint16_t* __restrict__ hb,
    const uint16_t* __restrict__ sb,
    const uint16_t* __restrict__ WihT, const uint16_t* __restrict__ WhhT,
    const uint16_t* __restrict__ WmemtT,
    const float* __restrict__ bias,
    const float* __restrict__ c_f32, const float* __restrict__ h_f32,
    float* __restrict__ out) {
  __shared__ uint16_t smem[2][2][64 * 32];  // [buffer][x/h (s reuses slot 0)]

  const int w     = threadIdx.x >> 5;
  const int mg    = w >> 2;                 // 0..1 -> 32-row slab in the 64-row tile
  const int m_off = mg * 32;
  const int n0    = blockIdx.y * 64 + (w & 3) * 16;
  const int mbase = blockIdx.x * 64;

  v8f accg[4][2] = {};   // [gate f,i,o,g][m-frag]
  v8f acct[2]    = {};   // memory-term GEMM

  // ---- gate GEMMs: K = 512 over x and h ----
  stage_tile(smem[0][0], xb + (size_t)mbase * IN, IN);
  stage_tile(smem[0][1], hb + (size_t)mbase * H, H);
  int buf = 0;
  for (int k = 0; k < IN; k += 32) {
    async_wait0();
    __syncthreads();
    if (k + 32 < IN) {
      stage_tile(smem[buf ^ 1][0], xb + (size_t)mbase * IN + k + 32, IN);
      stage_tile(smem[buf ^ 1][1], hb + (size_t)mbase * H + k + 32, H);
    }
    v16bf fx0 = load_frag(&smem[buf][0][(m_off + 0) * 32], 32);
    v16bf fx1 = load_frag(&smem[buf][0][(m_off + 16) * 32], 32);
    v16bf fh0 = load_frag(&smem[buf][1][(m_off + 0) * 32], 32);
    v16bf fh1 = load_frag(&smem[buf][1][(m_off + 16) * 32], 32);
#pragma unroll
    for (int gi = 0; gi < 4; ++gi) {
      v16bf bi = load_frag(WihT + (size_t)(gi * H + n0) * IN + k, IN);
      accg[gi][0] = __builtin_amdgcn_wmma_f32_16x16x32_bf16(false, fx0, false, bi, (short)0, accg[gi][0], false, false);
      accg[gi][1] = __builtin_amdgcn_wmma_f32_16x16x32_bf16(false, fx1, false, bi, (short)0, accg[gi][1], false, false);
      v16bf bh = load_frag(WhhT + (size_t)(gi * H + n0) * H + k, H);
      accg[gi][0] = __builtin_amdgcn_wmma_f32_16x16x32_bf16(false, fh0, false, bh, (short)0, accg[gi][0], false, false);
      accg[gi][1] = __builtin_amdgcn_wmma_f32_16x16x32_bf16(false, fh1, false, bh, (short)0, accg[gi][1], false, false);
    }
    __syncthreads();
    buf ^= 1;
  }

  // ---- memory-term GEMM: K = 1536 over s ----
  stage_tile(smem[0][0], sb + (size_t)mbase * OH, OH);
  buf = 0;
  const uint16_t* bt0 = WmemtT + (size_t)n0 * OH;
  for (int k = 0; k < OH; k += 32) {
    async_wait0();
    __syncthreads();
    if (k + 32 < OH) stage_tile(smem[buf ^ 1][0], sb + (size_t)mbase * OH + k + 32, OH);
    __builtin_prefetch(bt0 + k + 64, 0, 3);
    v16bf fs0 = load_frag(&smem[buf][0][(m_off + 0) * 32], 32);
    v16bf fs1 = load_frag(&smem[buf][0][(m_off + 16) * 32], 32);
    v16bf fb  = load_frag(bt0 + k, OH);
    acct[0] = __builtin_amdgcn_wmma_f32_16x16x32_bf16(false, fs0, false, fb, (short)0, acct[0], false, false);
    acct[1] = __builtin_amdgcn_wmma_f32_16x16x32_bf16(false, fs1, false, fb, (short)0, acct[1], false, false);
    __syncthreads();
    buf ^= 1;
  }

  // ---- fused LSTM epilogue ----
  const int lane = threadIdx.x & 31;
  const int col  = n0 + (lane & 15);
  const float bf = bias[col];
  const float bi = bias[H + col];
  const float bo = bias[2 * H + col];
  const float bg = bias[3 * H + col];
#pragma unroll
  for (int mf = 0; mf < 2; ++mf) {
#pragma unroll
    for (int v = 0; v < 8; ++v) {
      const int row    = mbase + m_off + mf * 16 + v + ((lane >> 4) << 3);
      const size_t idx = (size_t)row * H + col;
      const float cv = c_f32[idx];
      const float hv = h_f32[idx];
      const float f  = sigmoidf_fast(accg[0][mf][v] + bf);
      const float ii = sigmoidf_fast(accg[1][mf][v] + bi);
      const float o  = sigmoidf_fast(accg[2][mf][v] + bo);
      const float g  = tanhf(accg[3][mf][v] + bg);
      const float cn = f * cv + ii * g + acct[mf][v];
      const float hn = o * tanhf(cn);
      out[idx]                     = hn * MASK + hv * (1.0f - MASK);  // h_next
      out[(size_t)BATCH * H + idx] = cn * MASK + cv * (1.0f - MASK);  // c_next
    }
  }
}

// ---------------- host launcher ----------------

extern "C" void kernel_launch(void* const* d_in, const int* in_sizes, int n_in,
                              void* d_out, int out_size, void* d_ws, size_t ws_size,
                              hipStream_t stream) {
  const float* x        = (const float*)d_in[0];
  const float* h        = (const float*)d_in[1];
  const float* c        = (const float*)d_in[2];
  const float* me       = (const float*)d_in[3];
  const float* W_ih     = (const float*)d_in[4];
  const float* W_hh     = (const float*)d_in[5];
  const float* bias     = (const float*)d_in[6];
  const float* W_memi   = (const float*)d_in[7];
  const float* W_memh   = (const float*)d_in[8];
  const float* W_memt   = (const float*)d_in[9];
  const float* mem_bias = (const float*)d_in[10];
  float* out = (float*)d_out;

  uint8_t* ws = (uint8_t*)d_ws;
  size_t off = 0;
  auto take = [&](size_t elems) -> uint16_t* {
    uint16_t* p = (uint16_t*)(ws + off);
    off = (off + elems * sizeof(uint16_t) + 255) & ~(size_t)255;
    return p;
  };
  uint16_t* xb      = take((size_t)BATCH * IN);
  uint16_t* hb      = take((size_t)BATCH * H);
  uint16_t* meb     = take((size_t)BATCH * OH);
  uint16_t* sb      = take((size_t)BATCH * OH);
  uint16_t* WihT    = take((size_t)G4 * IN);
  uint16_t* WhhT    = take((size_t)G4 * H);
  uint16_t* WmemiT  = take((size_t)OH * IN);
  uint16_t* WmemhT  = take((size_t)OH * OH);
  uint16_t* WmemtT  = take((size_t)H * OH);

  const int TPB = 256;
  auto blocks = [](long long n, int t) { return (unsigned)((n + t - 1) / t); };

  // 1) activations -> bf16 (vectorized x4)
  cvt_bf16x4<<<blocks((long long)BATCH * IN / 4, TPB), TPB, 0, stream>>>(
      (const float4*)x, (uint2*)xb, BATCH * IN / 4);
  cvt_bf16x4<<<blocks((long long)BATCH * H / 4, TPB), TPB, 0, stream>>>(
      (const float4*)h, (uint2*)hb, BATCH * H / 4);
  cvt_bf16x4<<<blocks((long long)BATCH * OH / 4, TPB), TPB, 0, stream>>>(
      (const float4*)me, (uint2*)meb, BATCH * OH / 4);

  // 2) weights -> bf16, transposed to [N, K]
  cvt_transpose_bf16<<<blocks((long long)IN * G4, TPB), TPB, 0, stream>>>(W_ih,   WihT,   IN, G4);
  cvt_transpose_bf16<<<blocks((long long)H  * G4, TPB), TPB, 0, stream>>>(W_hh,   WhhT,   H,  G4);
  cvt_transpose_bf16<<<blocks((long long)IN * OH, TPB), TPB, 0, stream>>>(W_memi, WmemiT, IN, OH);
  cvt_transpose_bf16<<<blocks((long long)OH * OH, TPB), TPB, 0, stream>>>(W_memh, WmemhT, OH, OH);
  cvt_transpose_bf16<<<blocks((long long)OH * H,  TPB), TPB, 0, stream>>>(W_memt, WmemtT, OH, H);

  // 3) memory gate GEMM + fused sigmoid*me
  dim3 g1(BATCH / 128, OH / 64);
  memgate_gemm<<<g1, 256, 0, stream>>>(xb, meb, WmemiT, WmemhT, mem_bias, me, sb);

  // 4) gate GEMMs + memory-term GEMM + fused LSTM epilogue
  dim3 g2(BATCH / 64, H / 64);
  lstm_gemm<<<g2, 256, 0, stream>>>(xb, hb, sb, WihT, WhhT, WmemtT, bias, c, h, out);
}
